// GATLayer_69140383531301
// MI455X (gfx1250) — matrix operand order repaired and checked by reading, not compiled, not run on hardware
//
#include <hip/hip_runtime.h>

// ---------------- problem constants (match reference) ----------------
#define B_  4
#define N_  2048
#define FIN 128
#define H_  4
#define D_  64
#define HD  (H_ * D_)      // 256
#define ALPHA 0.2f
#define JT  32             // j-tile (WMMA K)
#define IT  128            // query rows per workgroup (8 waves x 16)
#define JPAD (JT + 8)      // 40 halves = 80B row stride: 16B-aligned chunks, conflict-free b32 reads

typedef __attribute__((ext_vector_type(16))) _Float16 v16h;
typedef __attribute__((ext_vector_type(8)))  float    v8f;

#define NEG_INF (-__builtin_inff())

// cross-lane helper (wave32; ds_bpermute index wraps on bits [6:2])
__device__ __forceinline__ float bperm_f(float v, int srcLane) {
    return __builtin_bit_cast(float,
        __builtin_amdgcn_ds_bpermute(srcLane << 2, __builtin_bit_cast(int, v)));
}

// CDNA5 async memory->LDS copy (16B per lane), tracked by ASYNCcnt.
// VDST = per-lane LDS byte address (low 32 bits of the generic pointer are the
// LDS offset per the flat-aperture mapping), VADDR = 64-bit global address.
__device__ __forceinline__ void async_copy_b128(void* lds_dst, const void* gsrc) {
    unsigned l = (unsigned)(size_t)lds_dst;
    asm volatile("global_load_async_to_lds_b128 %0, %1, off"
                 :: "v"(l), "v"(gsrc) : "memory");
}

__device__ __forceinline__ void wait_async0() {
#if __has_builtin(__builtin_amdgcn_s_wait_asynccnt)
    __builtin_amdgcn_s_wait_asynccnt(0);
#else
    asm volatile("s_wait_asynccnt 0x0" ::: "memory");
#endif
}

// ---------------------------------------------------------------------
// Kernel 1: h = x @ W  (8192x128)*(128x256), f16 WMMA, f32 accumulate.
// One wave computes one 16x16 tile.
// Output layout: h16[b][head][d][n]  (d-major so attention tiles are
// contiguous along j and can be async-copied straight into LDS).
// ---------------------------------------------------------------------
__global__ __launch_bounds__(32) void gat_gemm_h(const float* __restrict__ x,
                                                 const float* __restrict__ W,
                                                 _Float16* __restrict__ h16)
{
    const int lane = threadIdx.x & 31;
    const int hi   = lane >> 4;        // 0: K 0-7 & 16-23, 1: K 8-15 & 24-31
    const int il   = lane & 15;

    const int rowbase = blockIdx.x * 16;   // 0..8191 rows of x (b*N+n)
    const int colbase = blockIdx.y * 16;   // 0..255 cols (head*64+d)

    v8f acc = {};

    #pragma unroll
    for (int kb = 0; kb < FIN; kb += 32) {
        // A fragment: 16x32 f16.  lane holds row M=il, element e -> K:
        //   e<8 : K = 8*hi + e ;  e>=8 : K = 16 + 8*hi + (e-8)
        v16h af, bf;
        #pragma unroll
        for (int e = 0; e < 16; ++e) {
            int k = kb + ((e < 8) ? (8 * hi + e) : (16 + 8 * hi + (e - 8)));
            af[e] = (_Float16)x[(size_t)(rowbase + il) * FIN + k];
        }
        // B fragment: 32x16 f16. lane holds col n=il, element e -> K same as A
        #pragma unroll
        for (int e = 0; e < 16; ++e) {
            int k = kb + ((e < 8) ? (8 * hi + e) : (16 + 8 * hi + (e - 8)));
            bf[e] = (_Float16)W[(size_t)k * HD + colbase + il];
        }
        acc = __builtin_amdgcn_wmma_f32_16x16x32_f16(false, af, false, bf,
                                                     (short)0, acc, false, false);
    }

    // C layout: VGPR r, lane L -> row M = r + 8*hi, col N = il
    #pragma unroll
    for (int r = 0; r < 8; ++r) {
        int g = rowbase + r + 8 * hi;        // global row = b*N + n
        int b = g >> 11;                     // /2048
        int n = g & (N_ - 1);
        int c = colbase + il;                // head*64 + d
        int head = c >> 6;
        int d = c & 63;
        h16[((size_t)(b * H_ + head) * D_ + d) * N_ + n] = (_Float16)acc[r];
    }
}

// ---------------------------------------------------------------------
// Kernel 2: ei/ej = h . a1 / a2   (per b, head, n); h is [bh][d][n]
// ---------------------------------------------------------------------
__global__ __launch_bounds__(256) void gat_eij(const _Float16* __restrict__ h16,
                                               const float* __restrict__ a,
                                               float* __restrict__ ei,
                                               float* __restrict__ ej)
{
    int gid = blockIdx.x * 256 + threadIdx.x;        // B*H*N = 32768
    if (gid >= B_ * H_ * N_) return;
    int bh = gid >> 11;
    int n  = gid & (N_ - 1);
    const _Float16* hp = h16 + (size_t)bh * D_ * N_ + n;
    float e1 = 0.f, e2 = 0.f;
    #pragma unroll 8
    for (int d = 0; d < D_; ++d) {
        float hv = (float)hp[(size_t)d * N_];
        e1 += hv * a[d];
        e2 += hv * a[D_ + d];
    }
    ei[gid] = e1;
    ej[gid] = e2;
}

// ---------------------------------------------------------------------
// Kernel 3: fused masked-softmax attention + aggregation (flash style).
// grid = (N/IT, H, B), block = 256 (8 waves, 16 query rows each).
// j-tiles of 32 are double-buffered with GLOBAL_LOAD_ASYNC_TO_LDS_B128
// (ASYNCcnt); P is built directly in the WMMA A-fragment layout; online
// softmax uses ds_bpermute lane exchanges; 4x v_wmma per j-step.
// ---------------------------------------------------------------------
__global__ __launch_bounds__(256) void gat_attn(const _Float16* __restrict__ h16,
                                                const float* __restrict__ ei,
                                                const float* __restrict__ ej,
                                                const int* __restrict__ adj,
                                                float* __restrict__ out4)
{
    __shared__ _Float16 sh_hT[2][D_][JPAD];  // [buf][d][j], 80B row stride
    __shared__ int      sh_adj[2][IT][JT];   // [buf][i][j]
    __shared__ float    sh_ej[2][JT];

    const int tid  = threadIdx.x;
    const int lane = tid & 31;
    const int wave = tid >> 5;
    const int hi   = lane >> 4;
    const int il   = lane & 15;

    const int ibase = blockIdx.x * IT;
    const int head  = blockIdx.y;
    const int b     = blockIdx.z;
    const int bh    = b * H_ + head;

    const int i_state  = wave * 16 + il;        // row-in-tile owned by this lane
    const float ei_val = ei[(size_t)bh * N_ + ibase + i_state];

    const _Float16* hsrc  = h16 + (size_t)bh * D_ * N_;       // [d][n]
    const int*      asrc  = adj + ((size_t)b * N_ + ibase) * N_;
    const float*    ejsrc = ej + (size_t)bh * N_;

    // async staging decomposition (one 16B chunk per thread per issue):
    //   hT tile : 64 rows (d) x 32 halves = 256 chunks -> d = tid>>2, j0 = (tid&3)*8
    //   adj tile: 128 rows (i) x 32 ints  = 1024 chunks -> 4 per thread
    const int st_d  = tid >> 2;
    const int st_j0 = (tid & 3) * 8;

    float m = NEG_INF, l = 0.f;
    v8f acc0 = {}, acc1 = {}, acc2 = {}, acc3 = {};

    // ---- prologue: stage tile 0 into buffer 0 ----
    async_copy_b128(&sh_hT[0][st_d][st_j0], hsrc + (size_t)st_d * N_ + st_j0);
    #pragma unroll
    for (int q = 0; q < 4; ++q) {
        int c  = tid + q * 256;
        int i  = c >> 3;
        int j0 = (c & 7) * 4;
        async_copy_b128(&sh_adj[0][i][j0], asrc + (size_t)i * N_ + j0);
    }
    if (tid < JT) sh_ej[0][tid] = ejsrc[tid];

    const int NIT = N_ / JT;  // 64
    for (int it = 0; it < NIT; ++it) {
        const int buf = it & 1;
        wait_async0();       // this wave's async copies into `buf` are in LDS
        __syncthreads();     // all waves' copies are in LDS; prev compute done

        // ---- issue async staging for the next tile into buf^1 ----
        if (it + 1 < NIT) {
            const int jb1 = (it + 1) * JT;
            async_copy_b128(&sh_hT[buf ^ 1][st_d][st_j0],
                            hsrc + (size_t)st_d * N_ + jb1 + st_j0);
            #pragma unroll
            for (int q = 0; q < 4; ++q) {
                int c  = tid + q * 256;
                int i  = c >> 3;
                int j0 = (c & 7) * 4;
                async_copy_b128(&sh_adj[buf ^ 1][i][j0],
                                asrc + (size_t)i * N_ + jb1 + j0);
            }
            if (tid < JT) sh_ej[buf ^ 1][tid] = ejsrc[jb1 + tid];
            __builtin_prefetch(asrc + jb1 + JT, 0, 1);
        }

        // ---- logits in the A-fragment slot order ----
        float s[16];
        float mt = NEG_INF;
        #pragma unroll
        for (int e = 0; e < 16; ++e) {
            int j = (e < 8) ? (8 * hi + e) : (16 + 8 * hi + (e - 8));
            float v = ei_val + sh_ej[buf][j];
            v = (v >= 0.f) ? v : ALPHA * v;                 // leaky relu
            if (sh_adj[buf][i_state][j] == 0) v = NEG_INF;  // mask
            s[e] = v;
            mt = fmaxf(mt, v);
        }
        mt = fmaxf(mt, bperm_f(mt, lane ^ 16));             // row max across lane pair
        float m_new = fmaxf(m, mt);
        bool  dead  = (m_new == NEG_INF);

        float scale = dead ? 0.f : __expf(m - m_new);
        float psum = 0.f;
        v16h ap;
        #pragma unroll
        for (int e = 0; e < 16; ++e) {
            float p = dead ? 0.f : __expf(s[e] - m_new);
            psum += p;
            ap[e] = (_Float16)p;
        }
        psum += bperm_f(psum, lane ^ 16);
        l = l * scale + psum;
        m = m_new;

        // rescale accumulators: row of VGPR r is (r + 8*hi); state lives in lane r+8*hi
        #pragma unroll
        for (int r = 0; r < 8; ++r) {
            float rs = bperm_f(scale, r + 8 * hi);
            acc0[r] *= rs; acc1[r] *= rs; acc2[r] *= rs; acc3[r] *= rs;
        }

        // B fragments: 32x16 f16, lane col n = il, element e -> K same as A (table order)
        v16h b0, b1, b2, b3;
        #pragma unroll
        for (int e = 0; e < 16; ++e) {
            int k = (e < 8) ? (8 * hi + e) : (16 + 8 * hi + (e - 8));
            b0[e] = sh_hT[buf][ 0 + il][k];
            b1[e] = sh_hT[buf][16 + il][k];
            b2[e] = sh_hT[buf][32 + il][k];
            b3[e] = sh_hT[buf][48 + il][k];
        }
        acc0 = __builtin_amdgcn_wmma_f32_16x16x32_f16(false, ap, false, b0, (short)0, acc0, false, false);
        acc1 = __builtin_amdgcn_wmma_f32_16x16x32_f16(false, ap, false, b1, (short)0, acc1, false, false);
        acc2 = __builtin_amdgcn_wmma_f32_16x16x32_f16(false, ap, false, b2, (short)0, acc2, false, false);
        acc3 = __builtin_amdgcn_wmma_f32_16x16x32_f16(false, ap, false, b3, (short)0, acc3, false, false);
    }

    // epilogue: divide by softmax denom, write per-head output [bh][n][d]
    #pragma unroll
    for (int r = 0; r < 8; ++r) {
        float lr  = bperm_f(l, r + 8 * hi);
        float inv = (lr > 0.f) ? (1.f / lr) : 0.f;
        int gi = ibase + wave * 16 + r + 8 * hi;
        size_t base = (((size_t)bh * N_) + gi) * D_ + il;
        out4[base +  0] = acc0[r] * inv;
        out4[base + 16] = acc1[r] * inv;
        out4[base + 32] = acc2[r] * inv;
        out4[base + 48] = acc3[r] * inv;
    }
}

// ---------------------------------------------------------------------
// Kernel 4: mean over heads -> (B, N, D)
// ---------------------------------------------------------------------
__global__ __launch_bounds__(256) void gat_mean(const float* __restrict__ out4,
                                                float* __restrict__ out)
{
    int idx = blockIdx.x * 256 + threadIdx.x;       // B*N*D = 524288
    if (idx >= B_ * N_ * D_) return;
    int d = idx & 63;
    int n = (idx >> 6) & (N_ - 1);
    int b = idx >> 17;
    float s = 0.f;
    #pragma unroll
    for (int h = 0; h < H_; ++h)
        s += out4[(((size_t)(b * H_ + h) * N_) + n) * D_ + d];
    out[idx] = 0.25f * s;
}

// ---------------------------------------------------------------------
extern "C" void kernel_launch(void* const* d_in, const int* in_sizes, int n_in,
                              void* d_out, int out_size, void* d_ws, size_t ws_size,
                              hipStream_t stream)
{
    const float* x   = (const float*)d_in[0];
    const int*   adj = (const int*)d_in[1];
    const float* W   = (const float*)d_in[2];
    const float* a   = (const float*)d_in[3];
    float* out = (float*)d_out;

    // workspace layout
    char* ws = (char*)d_ws;
    size_t off = 0;
    _Float16* h16 = (_Float16*)(ws + off); off += (size_t)B_ * H_ * N_ * D_ * sizeof(_Float16); // 4 MB
    off = (off + 255) & ~(size_t)255;
    float* ei = (float*)(ws + off); off += (size_t)B_ * H_ * N_ * sizeof(float);                // 128 KB
    off = (off + 255) & ~(size_t)255;
    float* ej = (float*)(ws + off); off += (size_t)B_ * H_ * N_ * sizeof(float);                // 128 KB
    off = (off + 255) & ~(size_t)255;
    float* out4 = (float*)(ws + off);                                                           // 8 MB

    // 1) h = x@W  (f16 WMMA)
    gat_gemm_h<<<dim3((B_ * N_) / 16, HD / 16), dim3(32), 0, stream>>>(x, W, h16);
    // 2) ei/ej
    gat_eij<<<dim3((B_ * H_ * N_) / 256), dim3(256), 0, stream>>>(h16, a, ei, ej);
    // 3) fused masked softmax attention + aggregation (async double-buffered)
    gat_attn<<<dim3(N_ / IT, H_, B_), dim3(256), 0, stream>>>(h16, ei, ej, adj, out4);
    // 4) mean over heads
    gat_mean<<<dim3((B_ * N_ * D_ + 255) / 256), dim3(256), 0, stream>>>(out4, out);
}
